// SelfAttention_22986664968789
// MI455X (gfx1250) — matrix operand order repaired and checked by reading, not compiled, not run on hardware
//
#include <hip/hip_runtime.h>
#include <hip/hip_bf16.h>

typedef __attribute__((ext_vector_type(16))) _Float16 v16h;
typedef __attribute__((ext_vector_type(8)))  _Float16 v8h;
typedef __attribute__((ext_vector_type(8)))  float    v8f;
typedef __attribute__((ext_vector_type(4)))  float    v4f;
typedef __attribute__((ext_vector_type(4)))  unsigned int v4u;
typedef __attribute__((ext_vector_type(4)))  int      v4i;
typedef __attribute__((ext_vector_type(8)))  int      v8i;

#define B_DIM 4
#define C_DIM 256
#define N_DIM 4096
#define KQ    32
#define JTOT  320   // 32(q) + 32(k) + 256(v)

union AFrag { v16h v; v8h h[2]; _Float16 e[16]; };
union H8   { v8h h; _Float16 e[8]; };

// single file-scope dynamic LDS symbol shared by kernels
extern __shared__ unsigned char dyn_smem[];

#if defined(__has_builtin)
#if __has_builtin(__builtin_amdgcn_tensor_load_to_lds)
#define HAVE_TDM 1
#endif
#endif

static __device__ __forceinline__ v8f wmma_f16(const AFrag& a, const AFrag& b, v8f c) {
    return __builtin_amdgcn_wmma_f32_16x16x32_f16(false, a.v, false, b.v,
                                                  (short)0, c, false, false);
}

static __device__ __forceinline__ void wait_tensorcnt0() {
#if defined(__has_builtin)
#if __has_builtin(__builtin_amdgcn_s_wait_tensorcnt)
    __builtin_amdgcn_s_wait_tensorcnt(0);
#else
    asm volatile("s_wait_tensorcnt 0" ::: "memory");
#endif
#else
    asm volatile("s_wait_tensorcnt 0" ::: "memory");
#endif
}

static __device__ __forceinline__ void wait_tensorcnt1() {
#if defined(__has_builtin)
#if __has_builtin(__builtin_amdgcn_s_wait_tensorcnt)
    __builtin_amdgcn_s_wait_tensorcnt(1);
#else
    asm volatile("s_wait_tensorcnt 1" ::: "memory");
#endif
#else
    asm volatile("s_wait_tensorcnt 1" ::: "memory");
#endif
}

#if defined(HAVE_TDM)
// 1D TDM descriptor: copy 16 KB (4096 dwords) contiguous global -> LDS.
// D# group0: count=1, lds_addr, global_addr[56:0], type=2 ("image").
// D# group1: data_size=4B, tensor_dim0=4096, tensor_dim1=1, tile_dim0=4096,
//            tile_dim1=1, tensor_dim0_stride=4096.  Groups 2/3 unused (zero).
static __device__ __forceinline__ void tdm_load_16kb(const void* gsrc, void* lds_dst) {
    unsigned long long ga = (unsigned long long)(uintptr_t)gsrc;
    unsigned int lds_off =
        (unsigned int)(uintptr_t)(__attribute__((address_space(3))) const void*)lds_dst;
    v4u g0 = { 1u,
               lds_off,
               (unsigned int)(ga & 0xffffffffu),
               (unsigned int)((ga >> 32) & 0x1ffffffu) | (2u << 30) };
    v8i g1 = { (int)(2u << 16),          // workgroup_mask=0 | data_size=4B
               (int)(0x1000u << 16),     // tensor_dim0[15:0]=4096 -> bits 31:16
               (int)(1u << 16),          // tensor_dim0 hi=0 | tensor_dim1 lo=1
               (int)(0x1000u << 16),     // tensor_dim1 hi=0 | tile_dim0=4096
               (int)1,                   // tile_dim1=1, tile_dim2=0
               (int)4096,                // tensor_dim0_stride lo32
               0, 0 };
    v4i gz = { 0, 0, 0, 0 };
#if __clang_major__ >= 23
    v8i gz8 = { 0, 0, 0, 0, 0, 0, 0, 0 };
    __builtin_amdgcn_tensor_load_to_lds(g0, g1, gz, gz, gz8, 0);
#else
    __builtin_amdgcn_tensor_load_to_lds(g0, g1, gz, gz, 0);
#endif
}
#endif

// ---------------------------------------------------------------------------
// Kernel 0: concatenate + convert weights -> Wcat f16 [320][256], bcat f32[320]
// ---------------------------------------------------------------------------
__global__ __launch_bounds__(256)
void sa_prep_kernel(const float* __restrict__ Wq, const float* __restrict__ bq,
                    const float* __restrict__ Wk, const float* __restrict__ bk,
                    const float* __restrict__ Wv, const float* __restrict__ bv,
                    _Float16* __restrict__ Wcat, float* __restrict__ bcat) {
    int idx = blockIdx.x * 256 + threadIdx.x;
    if (idx < JTOT * C_DIM) {
        int j = idx >> 8;
        int c = idx & 255;
        float v;
        if (j < 32)       v = Wq[j * C_DIM + c];
        else if (j < 64)  v = Wk[(j - 32) * C_DIM + c];
        else              v = Wv[(j - 64) * C_DIM + c];
        Wcat[idx] = (_Float16)v;
    }
    if (idx < JTOT) {
        float bvv;
        if (idx < 32)      bvv = bq[idx];
        else if (idx < 64) bvv = bk[idx - 32];
        else               bvv = bv[idx - 64];
        bcat[idx] = bvv;
    }
}

// ---------------------------------------------------------------------------
// Kernel 1: fused projections.  D[n][j] = x^T(n,c) @ Wcat^T(c,j) + bcat[j]
// Outputs: qh [B][N][32] f16, kh [B][N][32] f16,
//          vPack [B][N/32][256][32] f16  (K-step-contiguous B-operand tiles)
// ---------------------------------------------------------------------------
__global__ __launch_bounds__(256)
void sa_proj_kernel(const float* __restrict__ x, const _Float16* __restrict__ Wcat,
                    const float* __restrict__ bcat,
                    _Float16* __restrict__ qh, _Float16* __restrict__ kh,
                    _Float16* __restrict__ vPack) {
    __shared__ __align__(16) _Float16 xs[64 * 32];    // x^T tile [n][c]
    __shared__ __align__(16) _Float16 wl[JTOT * 32];  // W tile   [j][c]

    const int b   = blockIdx.x / (N_DIM / 64);
    const int nb  = blockIdx.x % (N_DIM / 64);
    const int n0  = nb * 64;
    const int tid  = threadIdx.x;
    const int lane = tid & 31;
    const int wv   = tid >> 5;
    const int rb    = wv & 3;     // row block 0..3 (16 rows each)
    const int jhalf = wv >> 2;    // col-tile parity
    const int mloc  = lane & 15;
    const int half  = lane >> 4;

    v8f acc[10] = {};

    for (int kc = 0; kc < 8; ++kc) {
        const int c0 = kc * 32;
#pragma unroll
        for (int i = 0; i < 8; ++i) {
            int idx = tid + i * 256;
            int cc = idx >> 6, nn = idx & 63;
            xs[nn * 32 + cc] =
                (_Float16)x[((size_t)b * C_DIM + (c0 + cc)) * N_DIM + n0 + nn];
        }
#pragma unroll
        for (int i = 0; i < 40; ++i) {
            int idx = tid + i * 256;
            int j = idx >> 5, cc = idx & 31;
            wl[j * 32 + cc] = Wcat[j * C_DIM + c0 + cc];
        }
        __syncthreads();

        AFrag a;
        const v8h* ar = (const v8h*)(xs + (rb * 16 + mloc) * 32);
        a.h[0] = ar[half];
        a.h[1] = ar[2 + half];
#pragma unroll
        for (int i = 0; i < 10; ++i) {
            int jt = jhalf + 2 * i;  // 0..19
            AFrag bf;
            const v8h* br = (const v8h*)(wl + (jt * 16 + mloc) * 32);
            bf.h[0] = br[half];
            bf.h[1] = br[2 + half];
            acc[i] = wmma_f16(a, bf, acc[i]);
        }
        __syncthreads();
    }

    // epilogue: add bias, split into q/k/vPack, store f16
#pragma unroll
    for (int i = 0; i < 10; ++i) {
        int jt = jhalf + 2 * i;
        int j  = jt * 16 + mloc;
        float bias = bcat[j];
#pragma unroll
        for (int r = 0; r < 8; ++r) {
            int n = n0 + rb * 16 + r + 8 * half;
            _Float16 val = (_Float16)(acc[i][r] + bias);
            if (j < 32) {
                qh[((size_t)b * N_DIM + n) * KQ + j] = val;
            } else if (j < 64) {
                kh[((size_t)b * N_DIM + n) * KQ + (j - 32)] = val;
            } else {
                int nk = n >> 5, nl = n & 31;
                vPack[(((size_t)b * (N_DIM / 32) + nk) * C_DIM + (j - 64)) * 32 + nl] = val;
            }
        }
    }
}

// ---------------------------------------------------------------------------
// Kernel 2: fused energy + softmax.  One workgroup owns 16 query rows; the full
// 16x4096 fp32 energy slab lives in LDS (256 KB of the WGP's 320 KB).
// ---------------------------------------------------------------------------
__global__ __launch_bounds__(256)
void sa_attn_kernel(const _Float16* __restrict__ qh, const _Float16* __restrict__ kh,
                    float* __restrict__ attnOut) {
    float* es  = (float*)dyn_smem;          // 16 * 4096
    float* red = es + 16 * 4096;            // 16 * 16 reduction scratch

    const int b   = blockIdx.x / (N_DIM / 16);
    const int nb  = blockIdx.x % (N_DIM / 16);
    const int n0  = nb * 16;
    const int tid  = threadIdx.x;
    const int lane = tid & 31;
    const int wv   = tid >> 5;
    const int mloc = lane & 15;
    const int half = lane >> 4;

    AFrag a;
    {
        const v8h* qr = (const v8h*)(qh + ((size_t)b * N_DIM + n0 + mloc) * KQ);
        a.h[0] = qr[half];
        a.h[1] = qr[2 + half];
    }
    for (int ct = wv; ct < N_DIM / 16; ct += 8) {
        const v8h* kr = (const v8h*)(kh + ((size_t)b * N_DIM + ct * 16 + mloc) * KQ);
        if (ct + 8 < N_DIM / 16)
            __builtin_prefetch((const char*)kr + 8 * 16 * KQ * 2, 0, 0);
        AFrag bf;
        bf.h[0] = kr[half];
        bf.h[1] = kr[2 + half];
        v8f d = {};
        d = wmma_f16(a, bf, d);
#pragma unroll
        for (int r = 0; r < 8; ++r)
            es[(r + 8 * half) * N_DIM + ct * 16 + mloc] = d[r];
    }
    __syncthreads();

    const int row = tid >> 4;
    const int l16 = tid & 15;
    float* erow = es + row * N_DIM;

    float mx = -3.402823466e38f;
    for (int c = l16; c < N_DIM; c += 16) mx = fmaxf(mx, erow[c]);
    red[row * 16 + l16] = mx;
    __syncthreads();
#pragma unroll
    for (int i = 0; i < 16; ++i) mx = fmaxf(mx, red[row * 16 + i]);

    float s = 0.f;
    for (int c = l16; c < N_DIM; c += 16) {
        float e = __expf(erow[c] - mx);
        erow[c] = e;
        s += e;
    }
    __syncthreads();
    red[row * 16 + l16] = s;
    __syncthreads();
    s = 0.f;
#pragma unroll
    for (int i = 0; i < 16; ++i) s += red[row * 16 + i];
    float inv = 1.f / s;

    float* orow = attnOut + ((size_t)b * N_DIM + n0 + row) * N_DIM;
    for (int c = l16; c < N_DIM; c += 16) orow[c] = erow[c] * inv;
}

// ---------------------------------------------------------------------------
// Kernel 3: out^T[m][o] = sum_n attn[m][n] * vT[n][o]; then gamma*out + x.
// V tiles (16 KB contiguous) streamed by the Tensor Data Mover into a
// double-buffered LDS region: DMA of tile nk+1 overlaps attn staging and the
// 8 WMMAs of tile nk.  TDM ops are in-order per wave, so after issuing the
// next load, s_wait_tensorcnt<=1 guarantees the current tile has landed.
// ---------------------------------------------------------------------------
__global__ __launch_bounds__(256)
void sa_out_kernel(const float* __restrict__ attn, const _Float16* __restrict__ vPack,
                   const float* __restrict__ x, const float* __restrict__ gamma_p,
                   float* __restrict__ out) {
    _Float16* as_ = (_Float16*)dyn_smem;   // 64 x 32  attn tile [m][nloc]   (4 KB)
    _Float16* vs0 = as_ + 64 * 32;         // 256 x 32 v tile buf0          (16 KB)
    _Float16* vs1 = vs0 + C_DIM * 32;      // 256 x 32 v tile buf1          (16 KB)
    float*    os  = (float*)dyn_smem;      // 64 x 256 staging (reused, 64 KB)

    const int b   = blockIdx.x / (N_DIM / 64);
    const int mb  = blockIdx.x % (N_DIM / 64);
    const int m0  = mb * 64;
    const int tid  = threadIdx.x;
    const int lane = tid & 31;
    const int wv   = tid >> 5;
    const int mloc = lane & 15;
    const int half = lane >> 4;
    const int rb   = wv & 3;          // m row-block
    const int ob0  = (wv >> 2) * 8;   // first of 8 o-tiles for this wave

    v8f acc[8] = {};
    const _Float16* vbatch = vPack + (size_t)b * (N_DIM / 32) * (C_DIM * 32);

#if defined(HAVE_TDM)
    if (wv == 0) tdm_load_16kb(vbatch, vs0);          // prime the pipeline
#endif

    for (int nk = 0; nk < N_DIM / 32; ++nk) {
        const int nbase = nk * 32;
        _Float16* vs = (nk & 1) ? vs1 : vs0;

#if defined(HAVE_TDM)
        if (wv == 0) {
            if (nk + 1 < N_DIM / 32) {
                tdm_load_16kb(vbatch + (size_t)(nk + 1) * (C_DIM * 32),
                              (nk & 1) ? vs0 : vs1);  // DMA next tile
                wait_tensorcnt1();                     // current tile landed
            } else {
                wait_tensorcnt0();
            }
        }
#else
        {
            const _Float16* vsrc = vbatch + (size_t)nk * (C_DIM * 32);
#pragma unroll
            for (int i = 0; i < 4; ++i) {
                int c4 = tid + i * 256;               // 1024 x 16B
                ((v4u*)vs)[c4] = ((const v4u*)vsrc)[c4];
            }
        }
#endif
        // attn tile 64 x 32 fp32 -> f16: 8 floats/thread, one b128 LDS store
        {
            int ml = tid >> 2, c8 = tid & 3;
            const float* ap =
                attn + ((size_t)b * N_DIM + m0 + ml) * N_DIM + nbase + c8 * 8;
            v4f f0 = *(const v4f*)ap;
            v4f f1 = *(const v4f*)(ap + 4);
            H8 hh;
            hh.e[0] = (_Float16)f0.x; hh.e[1] = (_Float16)f0.y;
            hh.e[2] = (_Float16)f0.z; hh.e[3] = (_Float16)f0.w;
            hh.e[4] = (_Float16)f1.x; hh.e[5] = (_Float16)f1.y;
            hh.e[6] = (_Float16)f1.z; hh.e[7] = (_Float16)f1.w;
            *(v8h*)(as_ + ml * 32 + c8 * 8) = hh.h;
        }
        if (nk + 1 < N_DIM / 32 && tid < 64)
            __builtin_prefetch(
                attn + ((size_t)b * N_DIM + m0 + tid) * N_DIM + nbase + 32, 0, 0);
        __syncthreads();

        AFrag a;
        const v8h* ar = (const v8h*)(as_ + (rb * 16 + mloc) * 32);
        a.h[0] = ar[half];
        a.h[1] = ar[2 + half];
#pragma unroll
        for (int i = 0; i < 8; ++i) {
            int o = (ob0 + i) * 16 + mloc;
            AFrag bf;
            const v8h* br = (const v8h*)(vs + o * 32);
            bf.h[0] = br[half];
            bf.h[1] = br[2 + half];
            acc[i] = wmma_f16(a, bf, acc[i]);
        }
        __syncthreads();
    }

    // stage accumulators to LDS [m][o] so the global store is coalesced in m
#pragma unroll
    for (int i = 0; i < 8; ++i) {
        int o = (ob0 + i) * 16 + mloc;
#pragma unroll
        for (int r = 0; r < 8; ++r) {
            int ml = rb * 16 + r + 8 * half;
            os[ml * C_DIM + o] = acc[i][r];
        }
    }
    __syncthreads();

    const float g = gamma_p[0];
    for (int i = 0; i < 64; ++i) {
        int idx = tid + i * 256;
        int o = idx >> 6, ml = idx & 63;
        size_t gi = ((size_t)b * C_DIM + o) * N_DIM + m0 + ml;
        out[gi] = g * os[ml * C_DIM + o] + x[gi];
    }
}

// ---------------------------------------------------------------------------
extern "C" void kernel_launch(void* const* d_in, const int* in_sizes, int n_in,
                              void* d_out, int out_size, void* d_ws, size_t ws_size,
                              hipStream_t stream) {
    const float* x  = (const float*)d_in[0];
    const float* Wq = (const float*)d_in[1];
    const float* bq = (const float*)d_in[2];
    const float* Wk = (const float*)d_in[3];
    const float* bk = (const float*)d_in[4];
    const float* Wv = (const float*)d_in[5];
    const float* bv = (const float*)d_in[6];
    const float* gm = (const float*)d_in[7];

    float* out  = (float*)d_out;
    float* attn = out + (size_t)B_DIM * C_DIM * N_DIM;  // tuple: [out | attention]

    unsigned char* ws = (unsigned char*)d_ws;
    _Float16* Wcat  = (_Float16*)ws;                                   // 320*256*2
    float*    bcat  = (float*)(ws + 163840);                           // 320*4
    _Float16* qh    = (_Float16*)(ws + 165120);                        // B*N*32*2
    _Float16* kh    = (_Float16*)(ws + 165120 + 1048576);
    _Float16* vPack = (_Float16*)(ws + 165120 + 2 * 1048576);          // B*N*256*2

    sa_prep_kernel<<<(JTOT * C_DIM + 255) / 256, 256, 0, stream>>>(
        Wq, bq, Wk, bk, Wv, bv, Wcat, bcat);

    sa_proj_kernel<<<B_DIM * (N_DIM / 64), 256, 0, stream>>>(
        x, Wcat, bcat, qh, kh, vPack);

    size_t attn_lds = (size_t)16 * N_DIM * sizeof(float) + 1024;       // 263168 B
    sa_attn_kernel<<<B_DIM * (N_DIM / 16), 256, attn_lds, stream>>>(
        qh, kh, attn);

    size_t out_lds = (size_t)64 * C_DIM * sizeof(float);               // 65536 B
    sa_out_kernel<<<B_DIM * (N_DIM / 64), 256, out_lds, stream>>>(
        attn, vPack, x, gm, out);
}
